// deform_conv3_61529701482578
// MI455X (gfx1250) — matrix operand order repaired and checked by reading, not compiled, not run on hardware
//
#include <hip/hip_runtime.h>

#define Bdim 8
#define Cdim 192
#define Hdim 96
#define Wdim 96
#define HW (Hdim * Wdim)
#define BHW (Bdim * HW)
#define KS 216                    // padded bf16 (ushort) LDS stride: 432B, 16B-aligned,
                                  // 108 dwords -> gcd(44,64)=4 -> conflict-free frag rows
#define OSTR 65                   // padded f32 stride for output staging

typedef __attribute__((ext_vector_type(16))) __bf16 v16bf;
typedef __attribute__((ext_vector_type(8)))  float  v8f;
typedef __attribute__((ext_vector_type(4)))  int    v4i;

union Frag { v16bf v; unsigned int u[8]; };

__device__ __forceinline__ unsigned short f2bf(float f) {
    unsigned int u = __float_as_uint(f);
    u += 0x7FFFu + ((u >> 16) & 1u);          // round to nearest even
    return (unsigned short)(u >> 16);
}
__device__ __forceinline__ float bf2f(unsigned short h) {
    return __uint_as_float(((unsigned int)h) << 16);
}
__device__ __forceinline__ float bflo(unsigned int u) { return __uint_as_float(u << 16); }
__device__ __forceinline__ float bfhi(unsigned int u) { return __uint_as_float(u & 0xFFFF0000u); }
__device__ __forceinline__ unsigned int packbf(float lo, float hi) {
    return (unsigned int)f2bf(lo) | ((unsigned int)f2bf(hi) << 16);
}
__device__ __forceinline__ int iclamp(int v, int lo, int hi) {
    return v < lo ? lo : (v > hi ? hi : v);
}

// ---- async global->LDS copy (CDNA5 GLOBAL_LOAD_ASYNC_TO_LDS_B128, ASYNCcnt) ----
#if __has_builtin(__builtin_amdgcn_global_load_async_to_lds_b128)
#define HAS_ASYNC 1
__device__ __forceinline__ void copy16(const void* g, void* l) {
    __builtin_amdgcn_global_load_async_to_lds_b128((v4i*)g, (v4i*)l, 0, 0);
}
#else
#define HAS_ASYNC 0
__device__ __forceinline__ void copy16(const void* g, void* l) {
    *(uint4*)l = *(const uint4*)g;
}
#endif
__device__ __forceinline__ void async_wait0() {
#if HAS_ASYNC
#if __has_builtin(__builtin_amdgcn_s_wait_asynccnt)
    __builtin_amdgcn_s_wait_asynccnt(0);
#else
    asm volatile("s_wait_asynccnt 0x0" ::: "memory");
#endif
#endif
}

// ---------------------------------------------------------------------------
// 1) Tiled transpose + f32->bf16:  x[B,C,H,W] -> xt[B,H,W,C]
// ---------------------------------------------------------------------------
__global__ __launch_bounds__(256)
void transpose_x_kernel(const float* __restrict__ x, unsigned short* __restrict__ xt) {
    __shared__ unsigned short tile[64][66];
    int blk = blockIdx.x;
    int hwb = blk % 144;
    int t2  = blk / 144;
    int cb  = t2 % 3;
    int b   = t2 / 3;
    int c0  = cb * 64;
    int hw0 = hwb * 64;
    int t   = threadIdx.x;
    {
        int r  = t >> 2;
        int cc = (t & 3) * 16;
        const float* src = x + ((size_t)(b * Cdim + c0 + r)) * HW + hw0 + cc;
        #pragma unroll
        for (int i = 0; i < 16; ++i) tile[r][cc + i] = f2bf(src[i]);
    }
    __syncthreads();
    #pragma unroll 4
    for (int it = 0; it < 16; ++it) {
        int idx = it * 256 + t;
        int m = idx >> 6;
        int n = idx & 63;
        xt[((size_t)b * HW + hw0 + m) * Cdim + c0 + n] = tile[n][m];
    }
}

// ---------------------------------------------------------------------------
// 2) Pack conv weights: w_conv[o][c][i][j] -> wpk[tap][o][c]  (bf16)
// ---------------------------------------------------------------------------
__global__ __launch_bounds__(256)
void pack_wconv_kernel(const float* __restrict__ w, unsigned short* __restrict__ wp) {
    int i = blockIdx.x * 256 + threadIdx.x;
    if (i >= Cdim * Cdim * 9) return;
    int tap = i % 9;
    int t   = i / 9;
    int c   = t % Cdim;
    int o   = t / Cdim;
    wp[((size_t)(tap * Cdim + o)) * Cdim + c] = f2bf(w[i]);
}

// 3) Pack offset weights: w_offset[oc][c][i][j] -> wofp[tap][ocp(32)][c] (bf16, zero pad)
__global__ __launch_bounds__(256)
void pack_woffB_kernel(const float* __restrict__ w, unsigned short* __restrict__ wp) {
    int i = blockIdx.x * 256 + threadIdx.x;
    if (i >= 9 * 32 * Cdim) return;
    int c   = i % Cdim;
    int t   = i / Cdim;
    int ocp = t % 32;
    int tap = t / 32;
    float v = (ocp < 18) ? w[((size_t)(ocp * Cdim) + c) * 9 + tap] : 0.0f;
    wp[i] = f2bf(v);
}

// ---------------------------------------------------------------------------
// 4) Offset conv as WMMA GEMM: off[px][oc] = b[oc] + sum_tap,c xnb * woff
//    block = 128 threads (4 waves); tile = 64 px (M) x 32 padded oc (N)
// ---------------------------------------------------------------------------
__global__ __launch_bounds__(128)
void offset_gemm_kernel(const unsigned short* __restrict__ xt,    // [B,H,W,C] bf16
                        const unsigned short* __restrict__ wofp,  // [9][32][192] bf16
                        const float* __restrict__ boff,           // [18]
                        float* __restrict__ off) {                // [B,H,W,18]
    __shared__ __align__(16) unsigned short sA[64 * KS];
    __shared__ __align__(16) unsigned short sB[32 * KS];

    const int tid  = threadIdx.x;
    const int lane = tid & 31;
    const int wv   = tid >> 5;
    const int px0  = blockIdx.x * 64;
    const int b    = px0 / HW;
    const int hw0  = px0 % HW;

    v8f acc[2];
    acc[0] = (v8f){0, 0, 0, 0, 0, 0, 0, 0};
    acc[1] = (v8f){0, 0, 0, 0, 0, 0, 0, 0};

    const int arow  = wv * 16 + (lane & 15);
    const int ahalf = (lane >> 4) * 16;
    const int bhalf = (lane >> 4) * 16;
    const int ncol  = lane & 15;

    for (int tap = 0; tap < 9; ++tap) {
        const int ti = tap / 3 - 1;
        const int tj = tap % 3 - 1;
        // ---- stage A: 64 neighbor rows (zero-fill out-of-bounds) ----
        {
            int row = tid >> 1;
            int ch0 = (tid & 1) * 96;
            int hw  = hw0 + row;
            int h   = hw / Wdim, w = hw % Wdim;
            int hh  = h + ti,   ww = w + tj;
            unsigned short* d = sA + row * KS + ch0;
            if ((unsigned)hh < (unsigned)Hdim && (unsigned)ww < (unsigned)Wdim) {
                const unsigned short* s =
                    xt + ((size_t)b * HW + hh * Wdim + ww) * Cdim + ch0;
                #pragma unroll
                for (int i = 0; i < 96; i += 8) copy16(s + i, d + i);
            } else {
                uint4 z = {0, 0, 0, 0};
                #pragma unroll
                for (int i = 0; i < 96; i += 8) *(uint4*)(d + i) = z;
            }
        }
        // ---- stage B: 32 rows x 192 ----
        {
            int row = tid >> 2;
            int ch  = (tid & 3) * 48;
            const unsigned short* s = wofp + ((size_t)(tap * 32 + row)) * Cdim + ch;
            unsigned short*       d = sB + row * KS + ch;
            #pragma unroll
            for (int i = 0; i < 48; i += 8) copy16(s + i, d + i);
        }
        async_wait0();
        __syncthreads();

        #pragma unroll
        for (int ks = 0; ks < 6; ++ks) {
            const int k0 = ks * 32;
            Frag A;
            const unsigned short* ap = sA + arow * KS + k0;
            #pragma unroll
            for (int j = 0; j < 4; ++j) A.u[j] = *(const unsigned int*)(ap + ahalf + 2 * j);
            #pragma unroll
            for (int j = 0; j < 4; ++j) A.u[4 + j] = *(const unsigned int*)(ap + 8 + ahalf + 2 * j);
            Frag Bf[2];
            #pragma unroll
            for (int nt = 0; nt < 2; ++nt) {
                const unsigned short* bp = sB + (nt * 16 + ncol) * KS + k0 + bhalf;
                #pragma unroll
                for (int j = 0; j < 8; ++j) Bf[nt].u[j] = *(const unsigned int*)(bp + 2 * j);
            }
            acc[0] = __builtin_amdgcn_wmma_f32_16x16x32_bf16(
                false, A.v, false, Bf[0].v, (short)0, acc[0], false, false);
            acc[1] = __builtin_amdgcn_wmma_f32_16x16x32_bf16(
                false, A.v, false, Bf[1].v, (short)0, acc[1], false, false);
        }
        __syncthreads();
    }

    // ---- store (only oc < 18), rows are contiguous 18-float segments ----
    const int mbase = wv * 16 + ((lane >> 4) ? 8 : 0);
    #pragma unroll
    for (int nt = 0; nt < 2; ++nt) {
        int n = nt * 16 + ncol;
        if (n < 18) {
            float bias = boff[n];
            #pragma unroll
            for (int r = 0; r < 8; ++r) {
                int m = mbase + r;
                off[(size_t)(px0 + m) * 18 + n] = acc[nt][r] + bias;
            }
        }
    }
}

// ---------------------------------------------------------------------------
// 5) Main deformable GEMM + SiLU via WMMA bf16 (f32 acc)
//    block = 128 threads (4 waves); tile = 64 px (M) x 64 out-ch (N)
// ---------------------------------------------------------------------------
__global__ __launch_bounds__(128)
void deform_gemm_kernel(const unsigned short* __restrict__ xt,   // [B,H,W,C] bf16
                        const float* __restrict__ off,           // [B,H,W,18]
                        const unsigned short* __restrict__ wpk,  // [9][192][192] bf16
                        float* __restrict__ out) {               // [B,192,H,W]
    __shared__ __align__(16) unsigned short sS[64 * KS];   // sampled activations
    __shared__ __align__(16) unsigned short sW[64 * KS];   // tap weights

    const int tid  = threadIdx.x;
    const int lane = tid & 31;
    const int wv   = tid >> 5;
    const int px0  = blockIdx.x * 64;
    const int o0   = blockIdx.y * 64;
    const int b    = px0 / HW;
    const int hw0  = px0 % HW;

    v8f acc[4];
    #pragma unroll
    for (int i = 0; i < 4; ++i) acc[i] = (v8f){0, 0, 0, 0, 0, 0, 0, 0};

    const int arow  = wv * 16 + (lane & 15);
    const int ahalf = (lane >> 4) * 16;   // A: K offset 0/16 per lane half
    const int bhalf = (lane >> 4) * 16;   // B: K offset 0/16 per lane half
    const int ncol  = lane & 15;

    for (int tap = 0; tap < 9; ++tap) {
        const int ti = tap / 3 - 1;
        const int tj = tap % 3 - 1;

        // ---- stage this tap's weight slab first (async, overlaps sampling) ----
        {
            int row = tid >> 1;
            int h2  = (tid & 1) * 96;
            const unsigned short* s = wpk + ((size_t)(tap * Cdim + o0 + row)) * Cdim + h2;
            unsigned short*       d = sW + row * KS + h2;
            #pragma unroll
            for (int i = 0; i < 96; i += 8) copy16(s + i, d + i);
        }
        // ---- bilinear-sampled activations: sS[64 px][192 ch], packed pairs ----
        {
            int pxl = tid >> 1;            // 0..63
            int ch0 = (tid & 1) * 96;      // half of the channels
            int px  = px0 + pxl;
            int hw  = hw0 + pxl;
            int h   = hw / Wdim;
            int w   = hw % Wdim;
            float2 dyx = *(const float2*)(off + (size_t)px * 18 + 2 * tap);
            float y  = (float)(h + ti) + dyx.x;
            float xc = (float)(w + tj) + dyx.y;
            float y0f = floorf(y), x0f = floorf(xc);
            float wy = y - y0f, wx = xc - x0f;
            int y0 = (int)y0f, x0 = (int)x0f;
            int y1 = y0 + 1,   x1 = x0 + 1;
            bool vy0 = (unsigned)y0 < (unsigned)Hdim, vy1 = (unsigned)y1 < (unsigned)Hdim;
            bool vx0 = (unsigned)x0 < (unsigned)Wdim, vx1 = (unsigned)x1 < (unsigned)Wdim;
            float w00 = (vy0 && vx0) ? (1.f - wy) * (1.f - wx) : 0.f;
            float w01 = (vy0 && vx1) ? (1.f - wy) * wx         : 0.f;
            float w10 = (vy1 && vx0) ? wy * (1.f - wx)         : 0.f;
            float w11 = (vy1 && vx1) ? wy * wx                 : 0.f;
            int y0c = iclamp(y0, 0, Hdim - 1), y1c = iclamp(y1, 0, Hdim - 1);
            int x0c = iclamp(x0, 0, Wdim - 1), x1c = iclamp(x1, 0, Wdim - 1);
            const size_t base = (size_t)b * HW;
            const unsigned int* u00 = (const unsigned int*)
                (xt + (base + y0c * Wdim + x0c) * Cdim + ch0);
            const unsigned int* u01 = (const unsigned int*)
                (xt + (base + y0c * Wdim + x1c) * Cdim + ch0);
            const unsigned int* u10 = (const unsigned int*)
                (xt + (base + y1c * Wdim + x0c) * Cdim + ch0);
            const unsigned int* u11 = (const unsigned int*)
                (xt + (base + y1c * Wdim + x1c) * Cdim + ch0);
            unsigned int* dst = (unsigned int*)(sS + pxl * KS + ch0);
            #pragma unroll 8
            for (int q = 0; q < 48; ++q) {       // 48 packed pairs = 96 channels
                unsigned int a = u00[q], bq = u01[q], cq = u10[q], dq = u11[q];
                float lo = w00 * bflo(a) + w01 * bflo(bq) + w10 * bflo(cq) + w11 * bflo(dq);
                float hi = w00 * bfhi(a) + w01 * bfhi(bq) + w10 * bfhi(cq) + w11 * bfhi(dq);
                dst[q] = packbf(lo, hi);
            }
        }
        async_wait0();
        __syncthreads();

        // ---- 6 k-steps of wmma over C=192 for this tap ----
        #pragma unroll
        for (int ks = 0; ks < 6; ++ks) {
            const int k0 = ks * 32;
            Frag A;
            const unsigned short* ap = sS + arow * KS + k0;
            #pragma unroll
            for (int j = 0; j < 4; ++j) A.u[j] = *(const unsigned int*)(ap + ahalf + 2 * j);
            #pragma unroll
            for (int j = 0; j < 4; ++j) A.u[4 + j] = *(const unsigned int*)(ap + 8 + ahalf + 2 * j);
            Frag Bf[4];
            #pragma unroll
            for (int nt = 0; nt < 4; ++nt) {
                const unsigned short* bp = sW + (nt * 16 + ncol) * KS + k0 + bhalf;
                #pragma unroll
                for (int j = 0; j < 8; ++j) Bf[nt].u[j] = *(const unsigned int*)(bp + 2 * j);
            }
            #pragma unroll
            for (int nt = 0; nt < 4; ++nt)
                acc[nt] = __builtin_amdgcn_wmma_f32_16x16x32_bf16(
                    false, A.v, false, Bf[nt].v, (short)0, acc[nt], false, false);
        }
        __syncthreads();   // before next tap overwrites sS / sW
    }

    // ---- SiLU + transpose-stage through LDS, coalesced f32 stores ----
    float* stg = (float*)sS;                    // 64 x OSTR floats (16640 B < 27648 B)
    const int mbase = wv * 16 + ((lane >> 4) ? 8 : 0);
    #pragma unroll
    for (int nt = 0; nt < 4; ++nt) {
        #pragma unroll
        for (int r = 0; r < 8; ++r) {
            float a = acc[nt][r];
            float s = a / (1.0f + __expf(-a));  // SiLU
            stg[(mbase + r) * OSTR + nt * 16 + ncol] = s;
        }
    }
    __syncthreads();
    #pragma unroll 4
    for (int idx = tid; idx < 64 * 64; idx += 128) {
        int n = idx >> 6;    // out channel local
        int m = idx & 63;    // pixel local (contiguous hw)
        out[((size_t)(b * Cdim + o0 + n)) * HW + hw0 + m] = stg[m * OSTR + n];
    }
}

// ---------------------------------------------------------------------------
extern "C" void kernel_launch(void* const* d_in, const int* in_sizes, int n_in,
                              void* d_out, int out_size, void* d_ws, size_t ws_size,
                              hipStream_t stream) {
    (void)in_sizes; (void)n_in; (void)out_size;
    const float* x    = (const float*)d_in[0];   // [B,C,H,W]
    const float* woff = (const float*)d_in[1];   // [18,C,3,3]
    const float* boff = (const float*)d_in[2];   // [18]
    const float* wcv  = (const float*)d_in[3];   // [C,C,3,3]
    float* out = (float*)d_out;

    const size_t off_bytes = (size_t)BHW * 18 * 4;          //  5,308,416
    const size_t xt_bytes  = (size_t)Bdim * HW * Cdim * 2;  // 28,311,552
    const size_t wpk_bytes = (size_t)9 * Cdim * Cdim * 2;   //    663,552
    const size_t wof_bytes = (size_t)9 * 32 * Cdim * 2;     //    110,592
    if (ws_size < off_bytes + xt_bytes + wpk_bytes + wof_bytes) return;

    float*          off  = (float*)d_ws;
    unsigned short* xt   = (unsigned short*)((char*)d_ws + off_bytes);
    unsigned short* wpk  = (unsigned short*)((char*)d_ws + off_bytes + xt_bytes);
    unsigned short* wofp = (unsigned short*)((char*)d_ws + off_bytes + xt_bytes + wpk_bytes);

    transpose_x_kernel<<<Bdim * 3 * 144, 256, 0, stream>>>(x, xt);
    pack_wconv_kernel<<<(Cdim * Cdim * 9 + 255) / 256, 256, 0, stream>>>(wcv, wpk);
    pack_woffB_kernel<<<(9 * 32 * Cdim + 255) / 256, 256, 0, stream>>>(woff, wofp);
    offset_gemm_kernel<<<BHW / 64, 128, 0, stream>>>(xt, wofp, boff, off);
    deform_gemm_kernel<<<dim3(BHW / 64, Cdim / 64), 128, 0, stream>>>(xt, off, wpk, out);
}
